// RNNCRFTagger_79663053406785
// MI455X (gfx1250) — compile-verified, matrix operand order checked by reading, not run on hardware
//
#include <hip/hip_runtime.h>
#include <hip/hip_bf16.h>
#include <math.h>

// ---------------------------------------------------------------------------
// BiLSTM + FC + CRF NLL for MI455X (gfx1250, wave32, WMMA bf16 16x16x32)
//  - input projection fused into recurrent step (K = E + H = 768)
//  - 32-row M-tiles per wave: each B fragment feeds 2 WMMAs
//  - CRF forward pass: one persistent 1024-thread workgroup, alpha in LDS
// ---------------------------------------------------------------------------

typedef __bf16 bf16_t;
typedef __attribute__((ext_vector_type(16))) __bf16 v16bf;
typedef __attribute__((ext_vector_type(8)))  __bf16 v8bf;
typedef __attribute__((ext_vector_type(8)))  float  v8f;

constexpr int Bc = 64, Tc = 512, Ec = 256, Hc = 512, Lc = 64;
constexpr int G4H = 4 * Hc;   // 2048 gate width

#define DEV_INLINE __device__ __forceinline__

// A/B fragment for v_wmma_f32_16x16x32_bf16: lane holds row (M or N) = lane&15,
// K elements {kb..kb+7} U {kb+16..kb+23} with kb = (lane>=16)*8.
// `p` points at rowStart + k0 + kb; both chunks are 16B-aligned contiguous loads.
DEV_INLINE v16bf load_frag16(const bf16_t* p) {
  v8bf lo = *(const v8bf*)(p);
  v8bf hi = *(const v8bf*)(p + 16);
  v16bf r;
#pragma unroll
  for (int i = 0; i < 8; ++i) { r[i] = lo[i]; r[i + 8] = hi[i]; }
  return r;
}

DEV_INLINE v8f wmma_bf16(v16bf a, v16bf b, v8f c) {
  return __builtin_amdgcn_wmma_f32_16x16x32_bf16(false, a, false, b, (short)0, c,
                                                 false, false);
}

DEV_INLINE float sigmoidf_(float x) { return 1.0f / (1.0f + __expf(-x)); }

// ---------------------------------------------------------------------------
// Utility kernels
// ---------------------------------------------------------------------------
__global__ void f32_to_bf16_kernel(const float* __restrict__ src,
                                   bf16_t* __restrict__ dst, size_t n) {
  size_t i = (size_t)blockIdx.x * blockDim.x + threadIdx.x;
  size_t stride = (size_t)gridDim.x * blockDim.x;
  for (; i < n; i += stride) dst[i] = (bf16_t)src[i];
}

__global__ void zero_f32_kernel(float* __restrict__ p, size_t n) {
  size_t i = (size_t)blockIdx.x * blockDim.x + threadIdx.x;
  size_t stride = (size_t)gridDim.x * blockDim.x;
  for (; i < n; i += stride) p[i] = 0.0f;
}

__global__ void bias_sum_kernel(const float* __restrict__ bi_f,
                                const float* __restrict__ bh_f,
                                const float* __restrict__ bi_b,
                                const float* __restrict__ bh_b,
                                float* __restrict__ bias) {
  int i = blockIdx.x * blockDim.x + threadIdx.x;
  if (i < G4H) {
    bias[i]       = bi_f[i] + bh_f[i];
    bias[G4H + i] = bi_b[i] + bh_b[i];
  }
}

// ---------------------------------------------------------------------------
// One LSTM time step, both directions (blockIdx.z).  Fuses input projection
// (K over E) and recurrence (K over H).  Each wave: 32(b) x 16(h) tile for all
// four gates: 2 A fragments + 1 B fragment per gate -> 8 WMMAs per K chunk.
// grid = (H/64, B/32, 2), block = 128 (4 waves)
// ---------------------------------------------------------------------------
__global__ __launch_bounds__(128)
void lstm_step_kernel(const bf16_t* __restrict__ x_bf,   // [B][T][E]
                      const bf16_t* __restrict__ wih_bf, // [2][4H][E]
                      const bf16_t* __restrict__ whh_bf, // [2][4H][H]
                      const float*  __restrict__ bias,   // [2][4H]  (b_ih+b_hh)
                      const bf16_t* __restrict__ h_in,   // [2][B][H]
                      bf16_t*       __restrict__ h_out,  // [2][B][H]
                      float*        __restrict__ c_st,   // [2][B][H]
                      bf16_t*       __restrict__ hs,     // [2][T][B][H]
                      int step) {
  const int dir  = blockIdx.z;
  const int t    = dir ? (Tc - 1 - step) : step;
  const int wave = threadIdx.x >> 5;
  const int lane = threadIdx.x & 31;
  const int lrow = lane & 15;          // M (=b) / N (=h) index within tile
  const int kb   = (lane >> 4) * 8;    // K sub-offset per lane half
  const int b0   = blockIdx.y * 32;
  const int h0   = blockIdx.x * 64 + wave * 16;

  v8f acc[4][2];
#pragma unroll
  for (int g = 0; g < 4; ++g)
#pragma unroll
    for (int s = 0; s < 2; ++s) acc[g][s] = (v8f){};

  // ---- input projection part: K over E ----
  {
    const bf16_t* arow0 = x_bf + ((size_t)(b0 + lrow) * Tc + t) * Ec;
    const bf16_t* arow1 = x_bf + ((size_t)(b0 + 16 + lrow) * Tc + t) * Ec;
    const bf16_t* wrow  = wih_bf + (size_t)dir * G4H * Ec +
                          (size_t)(h0 + lrow) * Ec;
    const size_t gstr = (size_t)Hc * Ec;
#pragma unroll 2
    for (int k0 = 0; k0 < Ec; k0 += 32) {
      __builtin_prefetch(arow0 + k0 + 32, 0, 1);   // global_prefetch_b8
      v16bf a0 = load_frag16(arow0 + k0 + kb);
      v16bf a1 = load_frag16(arow1 + k0 + kb);
#pragma unroll
      for (int g = 0; g < 4; ++g) {
        v16bf w = load_frag16(wrow + g * gstr + k0 + kb);
        acc[g][0] = wmma_bf16(a0, w, acc[g][0]);
        acc[g][1] = wmma_bf16(a1, w, acc[g][1]);
      }
    }
  }

  // ---- recurrent part: K over H ----
  {
    const bf16_t* arow0 = h_in + ((size_t)dir * Bc + (b0 + lrow)) * Hc;
    const bf16_t* arow1 = h_in + ((size_t)dir * Bc + (b0 + 16 + lrow)) * Hc;
    const bf16_t* wrow  = whh_bf + (size_t)dir * G4H * Hc +
                          (size_t)(h0 + lrow) * Hc;
    const size_t gstr = (size_t)Hc * Hc;
#pragma unroll 2
    for (int k0 = 0; k0 < Hc; k0 += 32) {
      v16bf a0 = load_frag16(arow0 + k0 + kb);
      v16bf a1 = load_frag16(arow1 + k0 + kb);
#pragma unroll
      for (int g = 0; g < 4; ++g) {
        v16bf w = load_frag16(wrow + g * gstr + k0 + kb);
        acc[g][0] = wmma_bf16(a0, w, acc[g][0]);
        acc[g][1] = wmma_bf16(a1, w, acc[g][1]);
      }
    }
  }

  // ---- LSTM pointwise on accumulator registers ----
  // C/D layout: element r of v8f = (row M = 8*(lane>=16)+r, col N = lane&15)
  const int h  = h0 + (lane & 15);
  const float bI = bias[dir * G4H + 0 * Hc + h];
  const float bF = bias[dir * G4H + 1 * Hc + h];
  const float bG = bias[dir * G4H + 2 * Hc + h];
  const float bO = bias[dir * G4H + 3 * Hc + h];
#pragma unroll
  for (int s = 0; s < 2; ++s) {
#pragma unroll
    for (int r = 0; r < 8; ++r) {
      const int b = b0 + s * 16 + (lane >> 4) * 8 + r;
      const float gi = acc[0][s][r] + bI;
      const float gf = acc[1][s][r] + bF;
      const float gg = acc[2][s][r] + bG;
      const float go = acc[3][s][r] + bO;
      const size_t ch = ((size_t)dir * Bc + b) * Hc + h;
      const float cp = c_st[ch];
      const float cn = sigmoidf_(gf) * cp + sigmoidf_(gi) * tanhf(gg);
      const float hn = sigmoidf_(go) * tanhf(cn);
      c_st[ch]  = cn;
      h_out[ch] = (bf16_t)hn;
      hs[(((size_t)dir * Tc + t) * Bc + b) * Hc + h] = (bf16_t)hn;
    }
  }
}

// ---------------------------------------------------------------------------
// Emissions: em[b][t][l] = concat(hf,hb)[b][t][:] . fc_w[l][:] + fc_b[l]
// + 10000 on PAD column where label==PAD.   M=B*T, N=L=64, K=2H=1024.
// grid = (B*T/32), block = 128 (4 waves, each 16 cols x 32 rows)
// ---------------------------------------------------------------------------
__global__ __launch_bounds__(128)
void fc_emissions_kernel(const bf16_t* __restrict__ hs,     // [2][T][B][H]
                         const bf16_t* __restrict__ fcw,    // [L][2H]
                         const float*  __restrict__ fcb,    // [L]
                         const int*    __restrict__ labels, // [B][T]
                         float*        __restrict__ em) {   // [B][T][L]
  const int wave = threadIdx.x >> 5;
  const int lane = threadIdx.x & 31;
  const int lrow = lane & 15;
  const int kb   = (lane >> 4) * 8;
  const int m0 = blockIdx.x * 32;      // m = b*T + t ; 32 rows share b
  const int l0 = wave * 16;
  const int b  = m0 >> 9;              // T == 512
  const int tA = (m0 & 511) + lrow;    // this lane's A-row t (and +16)

  v8f acc0 = {}, acc1 = {};
  const bf16_t* wrow = fcw + (size_t)(l0 + lrow) * (2 * Hc);
#pragma unroll 4
  for (int k0 = 0; k0 < 2 * Hc; k0 += 32) {
    const int dir = k0 >> 9;
    const int kk  = k0 & 511;
    const bf16_t* abase =
        hs + (((size_t)dir * Tc + tA) * Bc + b) * Hc + kk + kb;
    v16bf a0 = load_frag16(abase);
    v16bf a1 = load_frag16(abase + (size_t)16 * Bc * Hc);  // t+16 rows
    v16bf w  = load_frag16(wrow + k0 + kb);
    acc0 = wmma_bf16(a0, w, acc0);
    acc1 = wmma_bf16(a1, w, acc1);
  }

  const int l = l0 + (lane & 15);
  const float bl = fcb[l];
#pragma unroll
  for (int s = 0; s < 2; ++s) {
#pragma unroll
    for (int r = 0; r < 8; ++r) {
      const int m = m0 + s * 16 + (lane >> 4) * 8 + r;
      const int t = m & 511;
      float v = (s ? acc1[r] : acc0[r]) + bl;
      if (l == 0 && labels[b * Tc + t] == 0) v += 10000.0f;   // PAD fix-up
      em[((size_t)b * Tc + t) * Lc + l] = v;
    }
  }
}

// ---------------------------------------------------------------------------
// CRF negative log-likelihood: single persistent workgroup (1024 thr, 32
// waves on one WGP), alpha + trans in LDS.  thread -> (b = tid>>4, 4 cols).
// ---------------------------------------------------------------------------
__global__ __launch_bounds__(1024)
void crf_nll_kernel(const float* __restrict__ em,      // [B][T][L]
                    const int*   __restrict__ labels,  // [B][T]
                    const float* __restrict__ startv,  // [L]
                    const float* __restrict__ endv,    // [L]
                    const float* __restrict__ trans,   // [L][L]
                    float* __restrict__ out) {
  __shared__ float sAlpha[Bc][Lc];
  __shared__ float sTrans[Lc][Lc];
  __shared__ float sRed[Bc];

  const int tid = threadIdx.x;
  for (int i = tid; i < Lc * Lc; i += blockDim.x)
    sTrans[i >> 6][i & 63] = trans[i];

  const int b  = tid >> 4;        // 0..63
  const int j0 = (tid & 15) * 4;  // 4 columns per thread
#pragma unroll
  for (int q = 0; q < 4; ++q)
    sAlpha[b][j0 + q] = startv[j0 + q] + em[((size_t)b * Tc) * Lc + j0 + q];
  __syncthreads();

  for (int t = 1; t < Tc; ++t) {
    const bool msk = labels[b * Tc + t] != 0;
    float mx[4] = {-1e30f, -1e30f, -1e30f, -1e30f};
    for (int i = 0; i < Lc; ++i) {
      const float a = sAlpha[b][i];
#pragma unroll
      for (int q = 0; q < 4; ++q) mx[q] = fmaxf(mx[q], a + sTrans[i][j0 + q]);
    }
    float s[4] = {0.f, 0.f, 0.f, 0.f};
    for (int i = 0; i < Lc; ++i) {
      const float a = sAlpha[b][i];
#pragma unroll
      for (int q = 0; q < 4; ++q) s[q] += __expf(a + sTrans[i][j0 + q] - mx[q]);
    }
    float v[4];
#pragma unroll
    for (int q = 0; q < 4; ++q)
      v[q] = mx[q] + __logf(s[q]) + em[((size_t)b * Tc + t) * Lc + j0 + q];
    __syncthreads();
    if (msk) {
#pragma unroll
      for (int q = 0; q < 4; ++q) sAlpha[b][j0 + q] = v[q];
    }
    __syncthreads();
  }

  if (tid < Bc) {
    const int bb = tid;
    // logZ
    float mx = -1e30f;
    for (int j = 0; j < Lc; ++j) mx = fmaxf(mx, sAlpha[bb][j] + endv[j]);
    float s = 0.f;
    for (int j = 0; j < Lc; ++j) s += __expf(sAlpha[bb][j] + endv[j] - mx);
    const float logZ = mx + __logf(s);
    // gold path score
    const int tg0 = labels[bb * Tc];
    float sc = startv[tg0] + em[((size_t)bb * Tc) * Lc + tg0];
    int cnt = (tg0 != 0) ? 1 : 0;
    for (int t = 1; t < Tc; ++t) {
      const int tg = labels[bb * Tc + t];
      const int tp = labels[bb * Tc + t - 1];
      if (tg != 0) {
        sc += sTrans[tp][tg] + em[((size_t)bb * Tc + t) * Lc + tg];
        ++cnt;
      }
    }
    int lastIdx = cnt - 1;
    if (lastIdx < 0) lastIdx = 0;
    sc += endv[labels[bb * Tc + lastIdx]];
    sRed[bb] = sc - logZ;
  }
  __syncthreads();
  if (tid == 0) {
    float tot = 0.f;
    for (int i = 0; i < Bc; ++i) tot += sRed[i];
    out[0] = -tot;
  }
}

// ---------------------------------------------------------------------------
// Host driver
// ---------------------------------------------------------------------------
extern "C" void kernel_launch(void* const* d_in, const int* in_sizes, int n_in,
                              void* d_out, int out_size, void* d_ws,
                              size_t ws_size, hipStream_t stream) {
  (void)in_sizes; (void)n_in; (void)out_size; (void)ws_size;

  const float* x    = (const float*)d_in[0];
  const int*   lab  = (const int*)d_in[1];
  const float* wihf = (const float*)d_in[2];
  const float* whhf = (const float*)d_in[3];
  const float* bihf = (const float*)d_in[4];
  const float* bhhf = (const float*)d_in[5];
  const float* wihb = (const float*)d_in[6];
  const float* whhb = (const float*)d_in[7];
  const float* bihb = (const float*)d_in[8];
  const float* bhhb = (const float*)d_in[9];
  const float* fcw  = (const float*)d_in[10];
  const float* fcb  = (const float*)d_in[11];
  const float* cst  = (const float*)d_in[12];
  const float* cen  = (const float*)d_in[13];
  const float* ctr  = (const float*)d_in[14];

  uint8_t* ws = (uint8_t*)d_ws;
  size_t off = 0;
  auto alloc = [&](size_t bytes) -> void* {
    void* p = ws + off;
    off = (off + bytes + 255) & ~(size_t)255;
    return p;
  };
  bf16_t* x_bf   = (bf16_t*)alloc((size_t)Bc * Tc * Ec * 2);   // 16 MB
  bf16_t* wih_bf = (bf16_t*)alloc((size_t)2 * G4H * Ec * 2);   //  2 MB
  bf16_t* whh_bf = (bf16_t*)alloc((size_t)2 * G4H * Hc * 2);   //  4 MB
  bf16_t* fcw_bf = (bf16_t*)alloc((size_t)Lc * 2 * Hc * 2);
  float*  bias   = (float*) alloc((size_t)2 * G4H * 4);
  bf16_t* hbufA  = (bf16_t*)alloc((size_t)2 * Bc * Hc * 2);
  bf16_t* hbufB  = (bf16_t*)alloc((size_t)2 * Bc * Hc * 2);
  float*  c_st   = (float*) alloc((size_t)2 * Bc * Hc * 4);
  bf16_t* hs     = (bf16_t*)alloc((size_t)2 * Tc * Bc * Hc * 2); // 64 MB
  float*  em     = (float*) alloc((size_t)Bc * Tc * Lc * 4);     //  8 MB

  auto cvt = [&](const float* s, bf16_t* d, size_t n) {
    int blocks = (int)((n + 255) / 256);
    if (blocks > 4096) blocks = 4096;
    f32_to_bf16_kernel<<<blocks, 256, 0, stream>>>(s, d, n);
  };
  cvt(x,    x_bf,                          (size_t)Bc * Tc * Ec);
  cvt(wihf, wih_bf,                        (size_t)G4H * Ec);
  cvt(wihb, wih_bf + (size_t)G4H * Ec,     (size_t)G4H * Ec);
  cvt(whhf, whh_bf,                        (size_t)G4H * Hc);
  cvt(whhb, whh_bf + (size_t)G4H * Hc,     (size_t)G4H * Hc);
  cvt(fcw,  fcw_bf,                        (size_t)Lc * 2 * Hc);

  bias_sum_kernel<<<(G4H + 255) / 256, 256, 0, stream>>>(bihf, bhhf, bihb,
                                                         bhhb, bias);
  // zero h0 (both double buffers, bf16 viewed as f32 words) and c0
  zero_f32_kernel<<<64, 256, 0, stream>>>((float*)hbufA,
                                          (size_t)2 * Bc * Hc / 2);
  zero_f32_kernel<<<64, 256, 0, stream>>>((float*)hbufB,
                                          (size_t)2 * Bc * Hc / 2);
  zero_f32_kernel<<<256, 256, 0, stream>>>(c_st, (size_t)2 * Bc * Hc);

  // 512 recurrent steps, both directions per launch, double-buffered h
  bf16_t* hin = hbufA;
  bf16_t* hout = hbufB;
  for (int s = 0; s < Tc; ++s) {
    lstm_step_kernel<<<dim3(Hc / 64, Bc / 32, 2), 128, 0, stream>>>(
        x_bf, wih_bf, whh_bf, bias, hin, hout, c_st, hs, s);
    bf16_t* tmp = hin; hin = hout; hout = tmp;
  }

  fc_emissions_kernel<<<dim3((Bc * Tc) / 32), 128, 0, stream>>>(
      hs, fcw_bf, fcb, lab, em);

  crf_nll_kernel<<<1, 1024, 0, stream>>>(em, lab, cst, cen, ctr,
                                         (float*)d_out);
}
